// GATModel_7413113553289
// MI455X (gfx1250) — matrix operand order repaired and checked by reading, not compiled, not run on hardware
//
#include <hip/hip_runtime.h>
#include <hip/hip_bf16.h>
#include <cstring>
#include <cstdint>

// ---------------------------------------------------------------------------
// Types for WMMA (CDNA5 gfx1250, wave32)
// ---------------------------------------------------------------------------
typedef __attribute__((ext_vector_type(16))) __bf16 v16bf;
typedef __attribute__((ext_vector_type(8)))  __bf16 v8bf;
typedef __attribute__((ext_vector_type(8)))  float  v8f;

#define NEG_SLOPE 0.2f
#define LN_EPS    1e-5f

// order-preserving float<->int encoding for atomicMax on floats
__device__ __forceinline__ int   fenc(float x){ int i = __float_as_int(x); return i < 0 ? (i ^ 0x7fffffff) : i; }
__device__ __forceinline__ float fdec(int i)  { return __int_as_float(i < 0 ? (i ^ 0x7fffffff) : i); }

// ---------------------------------------------------------------------------
// Prep / fill kernels
// ---------------------------------------------------------------------------
__global__ void k_f32_to_bf16(const float* __restrict__ in, __bf16* __restrict__ out, int n) {
  int i = blockIdx.x * blockDim.x + threadIdx.x;
  if (i < n) out[i] = (__bf16)in[i];
}

// W is [K, C] row-major; produce Wt[c*K + k] in bf16, zero-padded to Cpad cols
__global__ void k_transpose_bf16(const float* __restrict__ W, __bf16* __restrict__ Wt,
                                 int K, int C, int Cpad) {
  int i = blockIdx.x * blockDim.x + threadIdx.x;
  if (i >= Cpad * K) return;
  int c = i / K, k = i - c * K;
  float v = (c < C) ? W[(size_t)k * C + c] : 0.0f;
  Wt[i] = (__bf16)v;
}

__global__ void k_fill_f32(float* p, float v, int n) {
  int i = blockIdx.x * blockDim.x + threadIdx.x;
  if (i < n) p[i] = v;
}
__global__ void k_fill_i32(int* p, int v, int n) {
  int i = blockIdx.x * blockDim.x + threadIdx.x;
  if (i < n) p[i] = v;
}

// ---------------------------------------------------------------------------
// WMMA GEMM: C[M, ncols] = A[M, K](bf16) x Bt[ncols_pad, K](bf16, pre-transposed)
// One wave per 16x16 output tile, K stepped by 32 via v_wmma_f32_16x16x32_bf16.
// A lane layout (16-bit A 16x32): lanes 0-15 row M=r, K = {k0..k0+7, k0+16..k0+23};
//                                 lanes 16-31 row M=r, K shifted by +8.
// B lane layout: lane group g holds K = {k0+8g..+7, k0+16+8g..+7} for column n=r (same as A).
// C/D layout: VGPR v -> (M = v + 8g, N = r).
// ---------------------------------------------------------------------------
__global__ void k_gemm_bf16(const __bf16* __restrict__ A, const __bf16* __restrict__ Bt,
                            float* __restrict__ C, int K, int ncols, int nColTiles,
                            int totalWaves) {
  int wave = (blockIdx.x * blockDim.x + threadIdx.x) >> 5;
  if (wave >= totalWaves) return;
  int lane = threadIdx.x & 31;
  int rowTile = wave / nColTiles;
  int colTile = wave - rowTile * nColTiles;
  int g = lane >> 4;          // lane group (0 or 1)
  int r = lane & 15;

  const __bf16* arow = A  + (size_t)(rowTile * 16 + r) * K;
  const __bf16* brow = Bt + (size_t)(colTile * 16 + r) * K;

  v8f c = {0.f, 0.f, 0.f, 0.f, 0.f, 0.f, 0.f, 0.f};

  for (int k0 = 0; k0 < K; k0 += 32) {
    v8bf alo = *(const v8bf*)(arow + k0 + g * 8);        // K = k0+8g .. +7
    v8bf ahi = *(const v8bf*)(arow + k0 + 16 + g * 8);   // K = k0+16+8g .. +7
    v8bf blo = *(const v8bf*)(brow + k0 + g * 8);        // K = k0+8g .. +7
    v8bf bhi = *(const v8bf*)(brow + k0 + 16 + g * 8);   // K = k0+16+8g .. +7
    v16bf a, b;
#pragma unroll
    for (int i = 0; i < 8; ++i) {
      a[i] = alo[i]; a[i + 8] = ahi[i];
      b[i] = blo[i]; b[i + 8] = bhi[i];
    }
    c = __builtin_amdgcn_wmma_f32_16x16x32_bf16(false, a, false, b, (short)0, c,
                                                false, false);
  }

  int col = colTile * 16 + r;
  if (col < ncols) {
#pragma unroll
    for (int v = 0; v < 8; ++v) {
      int m = rowTile * 16 + v + 8 * g;
      C[(size_t)m * ncols + col] = c[v];
    }
  }
}

// ---------------------------------------------------------------------------
// Per-node attention coefficients: a_s[n,h] = sum_c h[n,h,c]*att_s[h,c]
// ---------------------------------------------------------------------------
__global__ void k_node_attn(const float* __restrict__ h,
                            const float* __restrict__ att_s, const float* __restrict__ att_d,
                            float* __restrict__ a_s, float* __restrict__ a_d,
                            int N, int H, int Cc) {
  int t = blockIdx.x * blockDim.x + threadIdx.x;
  if (t >= N * H) return;
  int n = t / H, hh = t - n * H;
  const float* hp = h + (size_t)n * H * Cc + hh * Cc;
  float ss = 0.f, dd = 0.f;
  for (int c = 0; c < Cc; ++c) { ss += hp[c] * att_s[hh * Cc + c]; dd += hp[c] * att_d[hh * Cc + c]; }
  a_s[t] = ss; a_d[t] = dd;
}

// ---------------------------------------------------------------------------
// Edge pass A: alpha = leaky_relu(a_s[src]+a_d[dst]); segment max over dst
// ---------------------------------------------------------------------------
__global__ void k_edge_alpha_max(const int* __restrict__ ei, int E, int EP, int H,
                                 const float* __restrict__ a_s, const float* __restrict__ a_d,
                                 float* __restrict__ ealpha, int* __restrict__ maxb) {
  int t = blockIdx.x * blockDim.x + threadIdx.x;
  if (t >= EP * H) return;
  int e = t / H, hh = t - e * H;
  int s, d;
  if (e < E) { s = ei[e]; d = ei[E + e]; } else { s = d = e - E; }
  float a = a_s[(size_t)s * H + hh] + a_d[(size_t)d * H + hh];
  a = (a > 0.f) ? a : NEG_SLOPE * a;
  ealpha[t] = a;
  atomicMax(&maxb[(size_t)d * H + hh], fenc(a));
}

// Edge pass B: w = exp(alpha - max[dst]); overwrite ealpha with w; sum denom
__global__ void k_edge_expsum(const int* __restrict__ ei, int E, int EP, int H,
                              float* __restrict__ ealpha, const int* __restrict__ maxb,
                              float* __restrict__ denom) {
  int t = blockIdx.x * blockDim.x + threadIdx.x;
  if (t >= EP * H) return;
  int e = t / H, hh = t - e * H;
  int d = (e < E) ? ei[E + e] : (e - E);
  float m = fdec(maxb[(size_t)d * H + hh]);
  float w = __expf(ealpha[t] - m);
  ealpha[t] = w;
  atomicAdd(&denom[(size_t)d * H + hh], w);
}

// Edge pass C: acc[dst, f] += h[src, f] * w / (denom[dst, h] + 1e-16)
__global__ void k_edge_scatter(const int* __restrict__ ei, int E, int EP, int H, int Cc,
                               const float* __restrict__ h, const float* __restrict__ ealpha,
                               const float* __restrict__ denom, float* __restrict__ acc) {
  int F = H * Cc;
  int t = blockIdx.x * blockDim.x + threadIdx.x;
  if (t >= EP * F) return;
  int e = t / F, f = t - e * F;
  int hh = f / Cc;
  int s, d;
  if (e < E) { s = ei[e]; d = ei[E + e]; } else { s = d = e - E; }
  float w = ealpha[(size_t)e * H + hh];
  float coef = w / (denom[(size_t)d * H + hh] + 1e-16f);
  atomicAdd(&acc[(size_t)d * F + f], h[(size_t)s * F + f] * coef);
}

// ---------------------------------------------------------------------------
// LayerNorm(acc + bias) -> ELU -> bf16 output (one wave32 per node)
// ---------------------------------------------------------------------------
__global__ void k_ln_elu_bf(const float* __restrict__ acc, const float* __restrict__ bias,
                            const float* __restrict__ gamma, const float* __restrict__ beta,
                            __bf16* __restrict__ out_bf, int N, int F) {
  int wave = (blockIdx.x * blockDim.x + threadIdx.x) >> 5;
  int lane = threadIdx.x & 31;
  if (wave >= N) return;
  const float* row = acc + (size_t)wave * F;
  int per = F >> 5;                 // 4 (F=128) or 2 (F=64)
  float v[4];
  float sum = 0.f;
  for (int i = 0; i < per; ++i) {
    int f = lane + 32 * i;
    v[i] = row[f] + bias[f];
    sum += v[i];
  }
  for (int o = 16; o > 0; o >>= 1) sum += __shfl_xor(sum, o, 32);
  float mu = sum / (float)F;
  float var = 0.f;
  for (int i = 0; i < per; ++i) { float dv = v[i] - mu; var += dv * dv; }
  for (int o = 16; o > 0; o >>= 1) var += __shfl_xor(var, o, 32);
  var /= (float)F;
  float rs = rsqrtf(var + LN_EPS);
  for (int i = 0; i < per; ++i) {
    int f = lane + 32 * i;
    float y = (v[i] - mu) * rs * gamma[f] + beta[f];
    y = (y > 0.f) ? y : (__expf(y) - 1.0f);           // ELU
    out_bf[(size_t)wave * F + f] = (__bf16)y;
  }
}

// ---------------------------------------------------------------------------
// Final: out = log_softmax(acc + b3) over the 2 classes
// ---------------------------------------------------------------------------
__global__ void k_final_lsm(const float* __restrict__ acc, const float* __restrict__ b3,
                            float* __restrict__ out, int N) {
  int n = blockIdx.x * blockDim.x + threadIdx.x;
  if (n >= N) return;
  float x0 = acc[(size_t)n * 2 + 0] + b3[0];
  float x1 = acc[(size_t)n * 2 + 1] + b3[1];
  float m = fmaxf(x0, x1);
  float l = logf(__expf(x0 - m) + __expf(x1 - m)) + m;
  out[(size_t)n * 2 + 0] = x0 - l;
  out[(size_t)n * 2 + 1] = x1 - l;
}

// ---------------------------------------------------------------------------
// Host-side launch
// ---------------------------------------------------------------------------
static int enc_host(float x) { int i; memcpy(&i, &x, 4); return i < 0 ? (i ^ 0x7fffffff) : i; }

extern "C" void kernel_launch(void* const* d_in, const int* in_sizes, int n_in,
                              void* d_out, int out_size, void* d_ws, size_t ws_size,
                              hipStream_t stream) {
  const float* x   = (const float*)d_in[0];
  const int*   ei  = (const int*)d_in[1];
  const float* W1  = (const float*)d_in[2];
  const float* as1 = (const float*)d_in[3];
  const float* ad1 = (const float*)d_in[4];
  const float* b1  = (const float*)d_in[5];
  const float* g1  = (const float*)d_in[6];
  const float* be1 = (const float*)d_in[7];
  const float* W2  = (const float*)d_in[8];
  const float* as2 = (const float*)d_in[9];
  const float* ad2 = (const float*)d_in[10];
  const float* b2  = (const float*)d_in[11];
  const float* g2  = (const float*)d_in[12];
  const float* be2 = (const float*)d_in[13];
  const float* W3  = (const float*)d_in[14];
  const float* as3 = (const float*)d_in[15];
  const float* ad3 = (const float*)d_in[16];
  const float* b3  = (const float*)d_in[17];
  float* out = (float*)d_out;

  const int N  = in_sizes[0] / 128;     // 50000
  const int E  = in_sizes[1] / 2;       // 800000
  const int EP = E + N;                 // edges + self loops

  // ---- workspace layout ----
  char* wp = (char*)d_ws;
  auto wsalloc = [&](size_t bytes) -> void* {
    void* p = (void*)wp;
    wp += (bytes + 255) & ~(size_t)255;
    return p;
  };
  __bf16* bfin  = (__bf16*)wsalloc((size_t)N * 128 * 2); // bf16 layer input
  __bf16* wt1   = (__bf16*)wsalloc(128 * 128 * 2);
  __bf16* wt2   = (__bf16*)wsalloc(64 * 128 * 2);
  __bf16* wt3   = (__bf16*)wsalloc(16 * 64 * 2);
  float*  h     = (float*)wsalloc((size_t)N * 128 * 4);  // GEMM out / gather source
  float*  acc   = (float*)wsalloc((size_t)N * 128 * 4);  // scatter accumulator
  float*  a_s   = (float*)wsalloc((size_t)N * 2 * 4);
  float*  a_d   = (float*)wsalloc((size_t)N * 2 * 4);
  int*    maxb  = (int*)wsalloc((size_t)N * 2 * 4);
  float*  denom = (float*)wsalloc((size_t)N * 2 * 4);
  float*  ealpha= (float*)wsalloc((size_t)EP * 2 * 4);

  const int TB = 256;
  auto nb = [](long long n) { return (unsigned)((n + 255) / 256); };
  const int ENC_NEG = enc_host(-1e30f);

  // ---- prep ----
  k_f32_to_bf16<<<nb((long long)N * 128), TB, 0, stream>>>(x, bfin, N * 128);
  k_transpose_bf16<<<nb(128 * 128), TB, 0, stream>>>(W1, wt1, 128, 128, 128);
  k_transpose_bf16<<<nb(64 * 128), TB, 0, stream>>>(W2, wt2, 128, 64, 64);
  k_transpose_bf16<<<nb(16 * 64), TB, 0, stream>>>(W3, wt3, 64, 2, 16);

  auto run_layer = [&](const __bf16* wt, int K, int H, int Cc,
                       const float* att_s, const float* att_d) {
    int F = H * Cc;
    int nColTiles = (F + 15) / 16;
    int totalWaves = (N / 16) * nColTiles;
    k_gemm_bf16<<<nb((long long)totalWaves * 32), TB, 0, stream>>>(
        bfin, wt, h, K, F, nColTiles, totalWaves);
    k_node_attn<<<nb((long long)N * H), TB, 0, stream>>>(h, att_s, att_d, a_s, a_d, N, H, Cc);
    k_fill_i32<<<nb((long long)N * H), TB, 0, stream>>>(maxb, ENC_NEG, N * H);
    k_fill_f32<<<nb((long long)N * H), TB, 0, stream>>>(denom, 0.f, N * H);
    k_fill_f32<<<nb((long long)N * F), TB, 0, stream>>>(acc, 0.f, N * F);
    k_edge_alpha_max<<<nb((long long)EP * H), TB, 0, stream>>>(ei, E, EP, H, a_s, a_d, ealpha, maxb);
    k_edge_expsum<<<nb((long long)EP * H), TB, 0, stream>>>(ei, E, EP, H, ealpha, maxb, denom);
    k_edge_scatter<<<nb((long long)EP * F), TB, 0, stream>>>(ei, E, EP, H, Cc, h, ealpha, denom, acc);
  };

  // ---- layer 1: 128 -> (2 heads x 64) = 128, LN + ELU ----
  run_layer(wt1, 128, 2, 64, as1, ad1);
  k_ln_elu_bf<<<nb((long long)N * 32), TB, 0, stream>>>(acc, b1, g1, be1, bfin, N, 128);

  // ---- layer 2: 128 -> (2 heads x 32) = 64, LN + ELU ----
  run_layer(wt2, 128, 2, 32, as2, ad2);
  k_ln_elu_bf<<<nb((long long)N * 32), TB, 0, stream>>>(acc, b2, g2, be2, bfin, N, 64);

  // ---- layer 3: 64 -> (1 head x 2) = 2, log_softmax ----
  run_layer(wt3, 64, 1, 2, as3, ad3);
  k_final_lsm<<<nb(N), TB, 0, stream>>>(acc, b3, out, N);

  (void)n_in; (void)out_size; (void)ws_size;
}
